// OneFormerSwinLayer_58059367907883
// MI455X (gfx1250) — matrix operand order repaired and checked by reading, not compiled, not run on hardware
//
#include <hip/hip_runtime.h>
#include <hip/hip_bf16.h>

typedef __attribute__((ext_vector_type(16))) _Float16 v16h;
typedef __attribute__((ext_vector_type(8)))  float    v8f;
typedef unsigned int uint;

// ---------------- constants ----------------
#define CC 96
#define SHIFT 4
#define HEADS 3
#define HD 32
#define NWIN 64            // tokens per window
#define MLPD 384
#define NWINDOWS 8192      // B * 32 * 32

// ws layout (in _Float16 elements)
#define OFF_WQT 0
#define OFF_WKT 9216
#define OFF_WVT 18432
#define OFF_WOT 27648
#define OFF_W1T 36864      // [384][96]
#define OFF_W2T 73728      // [96][384]

// ---------------- WMMA helpers ----------------
__device__ __forceinline__ v8f wmma16(v16h a, v16h b, v8f c) {
  return __builtin_amdgcn_wmma_f32_16x16x32_f16(
      false, a, false, b, (short)0, c, false, false);
}

// A fragment: 16x32 f16, row-major source with 'stride' halves per row.
// lane m = lane&15; halves[0..7]=A[m][k0 + (lane<16?0:8) + i], halves[8..15]=+16
__device__ __forceinline__ v16h load_fragA(const _Float16* base, int stride,
                                           int m0, int k0, int lane) {
  int m = m0 + (lane & 15);
  int ko = (lane < 16) ? 0 : 8;
  const _Float16* p0 = base + m * stride + k0 + ko;
  v16h a;
#pragma unroll
  for (int i = 0; i < 8; ++i) { a[i] = p0[i]; a[8 + i] = p0[16 + i]; }
  return a;
}

// B fragment: 32x16 f16 where B[k][n] = Bt[n][k], Bt row-major 'stride' halves.
// lane n = lane&15; halves[0..7] = Bt[n0+n][k0 + (lane<16?0:8) + i], halves[8..15]=+16
__device__ __forceinline__ v16h load_fragB(const _Float16* base, int stride,
                                           int n0, int k0, int lane) {
  int n = n0 + (lane & 15);
  int ko = (lane < 16) ? 0 : 8;
  const _Float16* p = base + n * stride + k0 + ko;
  v16h b;
#pragma unroll
  for (int i = 0; i < 8; ++i) { b[i] = p[i]; b[8 + i] = p[16 + i]; }
  return b;
}

// ---------------- kernel 0: weights -> f16 transposed ----------------
__global__ void prep_weights(const float* __restrict__ Wq, const float* __restrict__ Wk,
                             const float* __restrict__ Wv, const float* __restrict__ Wo,
                             const float* __restrict__ W1, const float* __restrict__ W2,
                             _Float16* __restrict__ ws) {
  int idx = blockIdx.x * blockDim.x + threadIdx.x;
  if (idx < 9216) {
    int n = idx / 96, k = idx % 96;
    ws[OFF_WQT + idx] = (_Float16)Wq[k * 96 + n];
  } else if (idx < 18432) {
    int e = idx - 9216; int n = e / 96, k = e % 96;
    ws[OFF_WKT + e] = (_Float16)Wk[k * 96 + n];
  } else if (idx < 27648) {
    int e = idx - 18432; int n = e / 96, k = e % 96;
    ws[OFF_WVT + e] = (_Float16)Wv[k * 96 + n];
  } else if (idx < 36864) {
    int e = idx - 27648; int n = e / 96, k = e % 96;
    ws[OFF_WOT + e] = (_Float16)Wo[k * 96 + n];
  } else if (idx < 73728) {
    int e = idx - 36864; int n = e / 96, k = e % 96;   // W1t[384][96]
    ws[OFF_W1T + e] = (_Float16)W1[k * MLPD + n];
  } else if (idx < 110592) {
    int e = idx - 73728; int n = e / 384, k = e % 384; // W2t[96][384]
    ws[OFF_W2T + e] = (_Float16)W2[k * 96 + n];
  }
}

// ---------------- fully fused Swin layer: one block = one window ----------------
__global__ __launch_bounds__(256) void swin_fused(
    const _Float16* __restrict__ wsw,
    const float* __restrict__ hs,          // input hidden_states (B, H*W, C)
    const float* __restrict__ ln1g, const float* __restrict__ ln1b,
    const float* __restrict__ bq, const float* __restrict__ bk,
    const float* __restrict__ bv, const float* __restrict__ bo,
    const float* __restrict__ table,
    const float* __restrict__ ln2g, const float* __restrict__ ln2b,
    const float* __restrict__ b1v, const float* __restrict__ b2v,
    float* __restrict__ out) {
  // 64140 bytes, phase-overlapped regions
  __shared__ __align__(16) char smem[64140];
  _Float16* Xs   = (_Float16*)(smem);            // 12KB  LN1 out; reused as Hs (LN2 out)
  _Float16* Qh   = (_Float16*)(smem + 12288);    // 4KB
  _Float16* Kh   = (_Float16*)(smem + 16384);    // 4KB
  _Float16* Vth  = (_Float16*)(smem + 20480);    // 4KB (transposed V)
  _Float16* Ctx  = (_Float16*)(smem + 24576);    // 12KB
  float*    Sc   = (float*)(smem + 36864);       // 16KB scores
  _Float16* P    = (_Float16*)(smem + 53248);    // 8KB  probs
  float*    Tb   = (float*)(smem + 61440);       // 2.7KB rel-bias table
  float*    Hid  = (float*)(smem + 36864);       // 24KB hidden f32 (reuses Sc+P)
  _Float16* Gs   = (_Float16*)(smem + 12288);    // 24KB gelu half (reuses QKV+Ctx)

  const int tid = threadIdx.x;
  const int lane = tid & 31;
  const int wave = tid >> 5;
  const int w = blockIdx.x;
  const int b = w >> 10, wy = (w >> 5) & 31, wx = w & 31;
  const int roff = (lane < 16) ? 0 : 8;
  const int nlo = lane & 15;

  const _Float16* Wqt = wsw + OFF_WQT;
  const _Float16* Wkt = wsw + OFF_WKT;
  const _Float16* Wvt = wsw + OFF_WVT;
  const _Float16* Wot = wsw + OFF_WOT;
  const _Float16* W1t = wsw + OFF_W1T;
  const _Float16* W2t = wsw + OFF_W2T;

  // ---- per-token global index for this window (token j = r*8+cw) ----
  // shifted coords -> source/destination coords (same map both directions)
  const int tok = tid >> 2;            // 0..63, 4 threads per token
  const int csub = (tid & 3) * 24;     // 24 channels per thread
  const int sy0 = (wy * 8 + (tok >> 3) + SHIFT) & 255;
  const int sx0 = (wx * 8 + (tok & 7) + SHIFT) & 255;
  const size_t gtok = (size_t)b * 65536 + (size_t)sy0 * 256 + sx0;

  // shortcut values for this thread's 24 (token, channel) slots: loaded once,
  // kept in registers across the attention phase (saves a 201MB re-read)
  float v[24];

  // ================= phase A0: LN1 (gathered, shifted) -> Xs f16 =================
  {
    const float* src = hs + gtok * CC + csub;
    float s = 0.f, ss = 0.f;
#pragma unroll
    for (int i = 0; i < 24; ++i) { float t = src[i]; v[i] = t; s += t; ss += t * t; }
    s += __shfl_xor(s, 1, 32); ss += __shfl_xor(ss, 1, 32);
    s += __shfl_xor(s, 2, 32); ss += __shfl_xor(ss, 2, 32);
    float mu = s * (1.f / 96.f);
    float var = ss * (1.f / 96.f) - mu * mu;
    float rs = rsqrtf(var + 1e-5f);
#pragma unroll
    for (int i = 0; i < 24; ++i)
      Xs[tok * CC + csub + i] = (_Float16)((v[i] - mu) * rs * ln1g[csub + i] + ln1b[csub + i]);
    for (int i = tid; i < 675; i += 256) Tb[i] = table[i];
  }
  __syncthreads();

  // ================= phase A1: per-head attention =================
  for (int h = 0; h < HEADS; ++h) {
    const int hd0 = h * HD;
    // QKV GEMMs: each wave does tile=wave (mt 0..3, nt 0..1) for q,k,v
    {
      const int mt = wave >> 1, nt = wave & 1;
#pragma unroll
      for (int sel = 0; sel < 3; ++sel) {
        const _Float16* Wt = (sel == 0) ? Wqt : (sel == 1) ? Wkt : Wvt;
        const float* bias = (sel == 0) ? bq : (sel == 1) ? bk : bv;
        v8f acc = {};
        for (int kt = 0; kt < 3; ++kt) {
          v16h a  = load_fragA(Xs, CC, mt * 16, kt * 32, lane);
          v16h bf = load_fragB(Wt, CC, hd0 + nt * 16, kt * 32, lane);
          acc = wmma16(a, bf, acc);
        }
        int cl = nt * 16 + nlo;
        float bc = bias[hd0 + cl];
        if (sel == 2) {
#pragma unroll
          for (int r2 = 0; r2 < 8; ++r2)
            Vth[cl * NWIN + mt * 16 + roff + r2] = (_Float16)(acc[r2] + bc);
        } else {
          _Float16* dstM = (sel == 0) ? Qh : Kh;
#pragma unroll
          for (int r2 = 0; r2 < 8; ++r2)
            dstM[(mt * 16 + roff + r2) * HD + cl] = (_Float16)(acc[r2] + bc);
        }
      }
    }
    __syncthreads();
    // scores = Q K^T * scale + rel bias + shift mask
    for (int job = wave; job < 16; job += 8) {
      int mt = job >> 2, nt = job & 3;
      v16h a  = load_fragA(Qh, HD, mt * 16, 0, lane);
      v16h bf = load_fragB(Kh, HD, nt * 16, 0, lane);
      v8f acc = {};
      acc = wmma16(a, bf, acc);
      int jcol = nt * 16 + nlo;
      int rj = jcol >> 3, cj = jcol & 7;
      int regj = ((wy == 31) ? ((rj < 4) ? 1 : 2) : 0) * 4 +
                 ((wx == 31) ? ((cj < 4) ? 1 : 2) : 0);
#pragma unroll
      for (int r2 = 0; r2 < 8; ++r2) {
        int i = mt * 16 + roff + r2;
        int ri = i >> 3, ci = i & 7;
        int regi = ((wy == 31) ? ((ri < 4) ? 1 : 2) : 0) * 4 +
                   ((wx == 31) ? ((ci < 4) ? 1 : 2) : 0);
        int rpi = (ri - rj + 7) * 15 + (ci - cj + 7);
        float vsc = acc[r2] * 0.17677669529663687f + Tb[rpi * 3 + h];
        if (regi != regj) vsc -= 100.0f;
        Sc[i * NWIN + jcol] = vsc;
      }
    }
    __syncthreads();
    // softmax rows (4 threads per row) -> P f16
    {
      int row = tid >> 2;
      int c0 = (tid & 3) * 16;
      float vv[16];
      float mx = -1e30f;
#pragma unroll
      for (int i2 = 0; i2 < 16; ++i2) { vv[i2] = Sc[row * NWIN + c0 + i2]; mx = fmaxf(mx, vv[i2]); }
      mx = fmaxf(mx, __shfl_xor(mx, 1, 32));
      mx = fmaxf(mx, __shfl_xor(mx, 2, 32));
      float sm = 0.f;
#pragma unroll
      for (int i2 = 0; i2 < 16; ++i2) { vv[i2] = expf(vv[i2] - mx); sm += vv[i2]; }
      sm += __shfl_xor(sm, 1, 32);
      sm += __shfl_xor(sm, 2, 32);
      float inv = 1.f / sm;
#pragma unroll
      for (int i2 = 0; i2 < 16; ++i2) P[row * NWIN + c0 + i2] = (_Float16)(vv[i2] * inv);
    }
    __syncthreads();
    // ctx = P @ V_h (one tile per wave)
    {
      int mt = wave >> 1, nt = wave & 1;
      v8f acc = {};
      for (int kt = 0; kt < 2; ++kt) {
        v16h a  = load_fragA(P, NWIN, mt * 16, kt * 32, lane);
        v16h bf = load_fragB(Vth, NWIN, nt * 16, kt * 32, lane);
        acc = wmma16(a, bf, acc);
      }
      int cl = hd0 + nt * 16 + nlo;
#pragma unroll
      for (int r2 = 0; r2 < 8; ++r2)
        Ctx[(mt * 16 + roff + r2) * CC + cl] = (_Float16)acc[r2];
    }
    __syncthreads();
  }

  // ===== phase A2: attn = Ctx @ Wo + bo  (kept in LDS, f32; shortcut added in B0) =====
  for (int job = wave; job < 24; job += 8) {
    int mt = job / 6, nt = job % 6;
    v8f acc = {};
    for (int kt = 0; kt < 3; ++kt) {
      v16h a  = load_fragA(Ctx, CC, mt * 16, kt * 32, lane);
      v16h bf = load_fragB(Wot, CC, nt * 16, kt * 32, lane);
      acc = wmma16(a, bf, acc);
    }
    int col = nt * 16 + nlo;
    float bc = bo[col];
#pragma unroll
    for (int r2 = 0; r2 < 8; ++r2) {
      int i = mt * 16 + roff + r2;
      Hid[i * CC + col] = acc[r2] + bc;   // overwrites Sc/P region (dead)
    }
  }
  __syncthreads();

  // ====== phase B0: hidden = shortcut(reg) + attn; LN2 -> Hs f16 (reuses Xs) ======
  {
    float s = 0.f, ss = 0.f;
#pragma unroll
    for (int i = 0; i < 24; ++i) {
      float x = Hid[tok * CC + csub + i] + v[i];   // add shortcut from registers
      Hid[tok * CC + csub + i] = x;                // keep hidden for final residual
      v[i] = x;
      s += x; ss += x * x;
    }
    s += __shfl_xor(s, 1, 32); ss += __shfl_xor(ss, 1, 32);
    s += __shfl_xor(s, 2, 32); ss += __shfl_xor(ss, 2, 32);
    float mu = s * (1.f / 96.f);
    float var = ss * (1.f / 96.f) - mu * mu;
    float rs = rsqrtf(var + 1e-5f);
#pragma unroll
    for (int i = 0; i < 24; ++i)
      Xs[tok * CC + csub + i] = (_Float16)((v[i] - mu) * rs * ln2g[csub + i] + ln2b[csub + i]);
  }
  __syncthreads();

  // ========== phase B1: MLP in two K-halves (Gs = 64x192 f16, reused LDS) =====
  v8f accJ[3] = {v8f{}, v8f{}, v8f{}};
  for (int half = 0; half < 2; ++half) {
    // GEMM1 (+bias, exact GELU) for columns [half*192, half*192+192)
    for (int job = wave; job < 48; job += 8) {
      int mt = job & 3, ntl = job >> 2;          // ntl 0..11 within half
      int n0 = half * 192 + ntl * 16;
      v8f acc = {};
      for (int kt = 0; kt < 3; ++kt) {
        v16h a  = load_fragA(Xs, CC, mt * 16, kt * 32, lane);
        v16h bf = load_fragB(W1t, CC, n0, kt * 32, lane);
        acc = wmma16(a, bf, acc);
      }
      int col = n0 + nlo;
      float bc = b1v[col];
#pragma unroll
      for (int r2 = 0; r2 < 8; ++r2) {
        float xg = acc[r2] + bc;
        float gl = 0.5f * xg * (1.0f + erff(xg * 0.70710678118654752f));
        Gs[(mt * 16 + roff + r2) * 192 + ntl * 16 + nlo] = (_Float16)gl;
      }
    }
    __syncthreads();
    // GEMM2 partial accumulation over this K-half (K=192 -> 6 WMMA steps)
    {
      int ji = 0;
      for (int job = wave; job < 24; job += 8, ++ji) {
        int mt = job / 6, nt = job % 6;
        for (int kt = 0; kt < 6; ++kt) {
          v16h a  = load_fragA(Gs, 192, mt * 16, kt * 32, lane);
          v16h bf = load_fragB(W2t, MLPD, nt * 16, half * 192 + kt * 32, lane);
          accJ[ji] = wmma16(a, bf, accJ[ji]);
        }
      }
    }
    __syncthreads();
  }

  // ===== phase B2: out = hidden + mlp + b2, scattered back (single write) =====
  {
    int ji = 0;
    for (int job = wave; job < 24; job += 8, ++ji) {
      int mt = job / 6, nt = job % 6;
      int col = nt * 16 + nlo;
      float bc = b2v[col];
#pragma unroll
      for (int r2 = 0; r2 < 8; ++r2) {
        int i = mt * 16 + roff + r2;
        int sy = (wy * 8 + (i >> 3) + SHIFT) & 255;
        int sx = (wx * 8 + (i & 7) + SHIFT) & 255;
        size_t gi = ((size_t)b * 65536 + (size_t)sy * 256 + sx) * CC + col;
        out[gi] = Hid[i * CC + col] + accJ[ji][r2] + bc;
      }
    }
  }
}

// ---------------- launcher ----------------
extern "C" void kernel_launch(void* const* d_in, const int* in_sizes, int n_in,
                              void* d_out, int out_size, void* d_ws, size_t ws_size,
                              hipStream_t stream) {
  const float* hs    = (const float*)d_in[0];
  const float* ln1g  = (const float*)d_in[1];
  const float* ln1b  = (const float*)d_in[2];
  const float* Wq    = (const float*)d_in[3];
  const float* bq    = (const float*)d_in[4];
  const float* Wk    = (const float*)d_in[5];
  const float* bk    = (const float*)d_in[6];
  const float* Wv    = (const float*)d_in[7];
  const float* bv    = (const float*)d_in[8];
  const float* table = (const float*)d_in[9];
  const float* Wo    = (const float*)d_in[10];
  const float* bo    = (const float*)d_in[11];
  const float* ln2g  = (const float*)d_in[12];
  const float* ln2b  = (const float*)d_in[13];
  const float* W1    = (const float*)d_in[14];
  const float* b1    = (const float*)d_in[15];
  const float* W2    = (const float*)d_in[16];
  const float* b2    = (const float*)d_in[17];
  float* out = (float*)d_out;

  _Float16* wsh = (_Float16*)d_ws;

  prep_weights<<<432, 256, 0, stream>>>(Wq, Wk, Wv, Wo, W1, W2, wsh);
  swin_fused<<<NWINDOWS, 256, 0, stream>>>(wsh, hs, ln1g, ln1b, bq, bk, bv, bo,
                                           table, ln2g, ln2b, b1, b2, out);
}
